// AttentionLayer_71399536328773
// MI455X (gfx1250) — matrix operand order repaired and checked by reading, not compiled, not run on hardware
//
#include <hip/hip_runtime.h>
#include <cstddef>

// ---------------------------------------------------------------- types -----
typedef _Float16 f16;
typedef _Float16 v16h __attribute__((ext_vector_type(16)));
typedef _Float16 v8h  __attribute__((ext_vector_type(8)));
typedef _Float16 v4h  __attribute__((ext_vector_type(4)));
typedef float    v8f  __attribute__((ext_vector_type(8)));
typedef float    v4f  __attribute__((ext_vector_type(4)));

#define B_   2
#define S_   2048
#define HID_ 1024
#define NH_  16
#define HD_  64

// ------------------------------------------------ WMMA fragment loaders -----
// 16-bit A 16x32 (MxK), lane l: row = l%16, half h = l/16; contiguous k order used here (B matches):
//   VGPR0..3 hold K = 32t + 16h + (0..7)     (contiguous 8 halfs)
//   VGPR4..7 hold K = 32t + 16h + 8 + (0..7) (contiguous 8 halfs)
__device__ __forceinline__ v16h load_a16(const f16* __restrict__ row, int kbase, int h) {
  v8h lo = *(const v8h*)(row + kbase + 16 * h);
  v8h hi = *(const v8h*)(row + kbase + 16 * h + 8);
  return __builtin_shufflevector(lo, hi, 0,1,2,3,4,5,6,7,8,9,10,11,12,13,14,15);
}
// 16-bit B 32x16 (KxN), lane l: col = l%16, half h = l/16:
//   16 contiguous K values starting at 32t + 16h  (column-contiguous source)
__device__ __forceinline__ v16h load_b16(const f16* __restrict__ col, int kbase, int h) {
  return *(const v16h*)(col + kbase + 16 * h);
}

// ------------------------------------------------ fp32->fp16 (vectorized) ---
__global__ void cvt_f32_f16(const float* __restrict__ src, f16* __restrict__ dst, int n4) {
  int i = blockIdx.x * blockDim.x + threadIdx.x;    // i indexes groups of 4
  if (i < n4) {
    v4f f = *(const v4f*)(src + (size_t)i * 4);
    v4h o;
    o[0] = (f16)f[0]; o[1] = (f16)f[1]; o[2] = (f16)f[2]; o[3] = (f16)f[3];
    *(v4h*)(dst + (size_t)i * 4) = o;
  }
}

// --------------------------------------------------- generic projection -----
// C[M,N] = A[M,K] * W[N,K]^T + bias   (A,W row-major, K contiguous on both)
// WG = 4 waves; each wave owns a 16x64 C tile; WG covers 16 rows x 256 cols.
__global__ __launch_bounds__(128)
void proj_gemm(const f16* __restrict__ A, const f16* __restrict__ W,
               const float* __restrict__ bias,
               f16* __restrict__ Cf16, float* __restrict__ Cf32,
               int M, int N, int K) {
  const int lane = threadIdx.x & 31;
  const int wave = threadIdx.x >> 5;
  const int h    = lane >> 4;
  const int ln   = lane & 15;
  const int row0 = blockIdx.x * 16;
  const int col0 = blockIdx.y * 256 + wave * 64;

  const f16* arow = A + (size_t)(row0 + ln) * K;
  v8f acc[4] = {};

  for (int k = 0; k < K; k += 32) {
    v16h a = load_a16(arow, k, h);
#pragma unroll
    for (int t = 0; t < 4; ++t) {
      const f16* wrow = W + (size_t)(col0 + t * 16 + ln) * K;
      v16h b = load_b16(wrow, k, h);
      acc[t] = __builtin_amdgcn_wmma_f32_16x16x32_f16(false, a, false, b,
                                                      (short)0, acc[t], false, false);
    }
  }
  // C/D layout: lane l -> N = col0 + t*16 + l%16 ; VGPR r -> M = row0 + r + 8h
#pragma unroll
  for (int t = 0; t < 4; ++t) {
    const int n = col0 + t * 16 + ln;
    const float bv = bias[n];
#pragma unroll
    for (int r = 0; r < 8; ++r) {
      const int m = row0 + r + 8 * h;
      const float v = acc[t][r] + bv;
      if (Cf16) Cf16[(size_t)m * N + n] = (f16)v;
      else      Cf32[(size_t)m * N + n] = v;
    }
  }
}

// ------------------------------------------- V -> V^T per head [HD, S] ------
__global__ void v_transpose(const f16* __restrict__ Vin, f16* __restrict__ VT) {
  int i = blockIdx.x * blockDim.x + threadIdx.x;   // ((b*NH+hh)*HD + d)*S + s
  if (i >= B_ * NH_ * HD_ * S_) return;
  int s    = i & (S_ - 1);
  int rest = i >> 11;
  int d    = rest & (HD_ - 1);
  int bh   = rest >> 6;
  int hh   = bh & (NH_ - 1);
  int b    = bh >> 4;
  VT[i] = Vin[((size_t)(b * S_ + s)) * HID_ + hh * HD_ + d];
}

// -------------------------- scores (QK^T * sqrt(HD)) + causal softmax -------
// One WG per (b, head, 16-row block). 4 waves fill 16x2048 f32 scores in LDS
// (causal tiles only), then cooperative row softmax, then fully-coalesced
// b128 store of normalized probabilities (zeros above diagonal) to d_out.
__global__ __launch_bounds__(128)
void scores_softmax(const f16* __restrict__ Qt, const f16* __restrict__ Kt,
                    float* __restrict__ attn) {
  extern __shared__ float smem[];          // 16 * S_ floats = 128 KB
  __shared__ float red[16][8];
  __shared__ float rowval[16];

  const int qb = blockIdx.x, hh = blockIdx.y, b = blockIdx.z;
  const int lane = threadIdx.x & 31, wave = threadIdx.x >> 5;
  const int h = lane >> 4, ln = lane & 15;
  const int q0 = qb * 16;

  const f16* qrow = Qt + ((size_t)(b * S_ + q0 + ln)) * HID_ + hh * HD_;
  v16h a0 = load_a16(qrow, 0, h);
  v16h a1 = load_a16(qrow, 32, h);

  for (int nt = wave; nt <= qb; nt += 4) {          // causal: skip tiles > diag
    const f16* krow = Kt + ((size_t)(b * S_ + nt * 16 + ln)) * HID_ + hh * HD_;
    v8f acc = {};
    acc = __builtin_amdgcn_wmma_f32_16x16x32_f16(false, a0, false, load_b16(krow, 0, h),
                                                 (short)0, acc, false, false);
    acc = __builtin_amdgcn_wmma_f32_16x16x32_f16(false, a1, false, load_b16(krow, 32, h),
                                                 (short)0, acc, false, false);
#pragma unroll
    for (int r = 0; r < 8; ++r) {
      const int m   = r + 8 * h;
      const int col = nt * 16 + ln;
      float v = acc[r] * 8.0f;                      // / HD^-0.5 == * sqrt(64)
      if (col > q0 + m) v = -__builtin_inff();
      smem[m * S_ + col] = v;
    }
  }
  __syncthreads();

  // row softmax: 8 threads per row
  const int r = threadIdx.x >> 3, j = threadIdx.x & 7;
  const int q = q0 + r, ncols = q + 1;
  float mx = -__builtin_inff();
  for (int c = j; c < ncols; c += 8) mx = fmaxf(mx, smem[r * S_ + c]);
  red[r][j] = mx;
  __syncthreads();
  if (j == 0) {
    float m2 = red[r][0];
#pragma unroll
    for (int t = 1; t < 8; ++t) m2 = fmaxf(m2, red[r][t]);
    rowval[r] = m2;
  }
  __syncthreads();
  const float rmax = rowval[r];
  float sum = 0.f;
  for (int c = j; c < ncols; c += 8) {
    float e = __expf(smem[r * S_ + c] - rmax);
    smem[r * S_ + c] = e;
    sum += e;
  }
  red[r][j] = sum;
  __syncthreads();
  if (j == 0) {
    float s2 = 0.f;
#pragma unroll
    for (int t = 0; t < 8; ++t) s2 += red[r][t];
    rowval[r] = 1.0f / s2;
  }
  __syncthreads();

  // coalesced b128 write of 16 full rows — the dominant traffic of the layer
  float* base = attn + (((size_t)(b * NH_ + hh)) * S_ + q0) * S_;
  for (int rr = 0; rr < 16; ++rr) {
    const int qq  = q0 + rr;
    const float inv = rowval[rr];
    const float* srow = smem + rr * S_;
    for (int c = threadIdx.x * 4; c < S_; c += 512) {
      v4f v;
      v[0] = (c     <= qq) ? srow[c]     * inv : 0.0f;
      v[1] = (c + 1 <= qq) ? srow[c + 1] * inv : 0.0f;
      v[2] = (c + 2 <= qq) ? srow[c + 2] * inv : 0.0f;
      v[3] = (c + 3 <= qq) ? srow[c + 3] * inv : 0.0f;
      *(v4f*)(base + (size_t)rr * S_ + c) = v;
    }
  }
}

// ----------------------------- ctx = attn @ V  (per head, causal k-loop) ----
// One WG per (b, head, 16-row block); wave w owns 16-wide head-dim tile w.
__global__ __launch_bounds__(128)
void ctx_gemm(const float* __restrict__ attn, const f16* __restrict__ VT,
              f16* __restrict__ ctxh) {
  const int qb = blockIdx.x, hh = blockIdx.y, b = blockIdx.z;
  const int lane = threadIdx.x & 31, wave = threadIdx.x >> 5;
  const int h = lane >> 4, ln = lane & 15;
  const int q0 = qb * 16;

  const float* arow = attn + ((((size_t)(b * NH_ + hh)) * S_) + q0 + ln) * S_;
  const f16*   vcol = VT + (((size_t)(b * NH_ + hh)) * HD_ + wave * 16 + ln) * (size_t)S_;

  const int ktiles = ((q0 + 15) >> 5) + 1;          // causal extent (rest is 0)
  v8f acc = {};
  for (int kt = 0; kt < ktiles; ++kt) {
    const int kb = kt * 32;
    v8f f0 = *(const v8f*)(arow + kb + 16 * h);
    v8f f1 = *(const v8f*)(arow + kb + 16 * h + 8);
    v16h a;
#pragma unroll
    for (int i = 0; i < 8; ++i) { a[i] = (f16)f0[i]; a[8 + i] = (f16)f1[i]; }
    v16h bm = *(const v16h*)(vcol + kb + 16 * h);
    acc = __builtin_amdgcn_wmma_f32_16x16x32_f16(false, a, false, bm,
                                                 (short)0, acc, false, false);
  }
#pragma unroll
  for (int r = 0; r < 8; ++r) {
    const int q = q0 + r + 8 * h;
    ctxh[((size_t)(b * S_ + q)) * HID_ + hh * HD_ + wave * 16 + ln] = (f16)acc[r];
  }
}

// --------------------------------------------------------------- launch -----
extern "C" void kernel_launch(void* const* d_in, const int* in_sizes, int n_in,
                              void* d_out, int out_size, void* d_ws, size_t ws_size,
                              hipStream_t stream) {
  const float* x  = (const float*)d_in[0];
  const float* Wq = (const float*)d_in[1];
  const float* bq = (const float*)d_in[2];
  const float* Wk = (const float*)d_in[3];
  const float* bk = (const float*)d_in[4];
  const float* Wv = (const float*)d_in[5];
  const float* bv = (const float*)d_in[6];
  const float* Wo = (const float*)d_in[7];
  const float* bo = (const float*)d_in[8];

  float* out  = (float*)d_out;                       // [B,S,HID]
  float* attn = out + (size_t)B_ * S_ * HID_;        // [B,NH,S,S]

  const size_t nX = (size_t)B_ * S_ * HID_;          // 4,194,304
  const size_t nW = (size_t)HID_ * HID_;             // 1,048,576

  // workspace carve (all f16): ~56 MB total
  f16* xh   = (f16*)d_ws;
  f16* Wqh  = xh  + nX;
  f16* Wkh  = Wqh + nW;
  f16* Wvh  = Wkh + nW;
  f16* Woh  = Wvh + nW;
  f16* Qt   = Woh + nW;   // [B,S,HID]
  f16* Kt   = Qt  + nX;
  f16* Vt   = Kt  + nX;
  f16* VTh  = Vt  + nX;   // [B,NH,HD,S]
  f16* ctxh = VTh + nX;   // [B,S,HID]

  // 1) fp32 -> fp16 (b128-vectorized)
  const int nX4 = (int)(nX / 4), nW4 = (int)(nW / 4);
  cvt_f32_f16<<<(nX4 + 255) / 256, 256, 0, stream>>>(x,  xh,  nX4);
  cvt_f32_f16<<<(nW4 + 255) / 256, 256, 0, stream>>>(Wq, Wqh, nW4);
  cvt_f32_f16<<<(nW4 + 255) / 256, 256, 0, stream>>>(Wk, Wkh, nW4);
  cvt_f32_f16<<<(nW4 + 255) / 256, 256, 0, stream>>>(Wv, Wvh, nW4);
  cvt_f32_f16<<<(nW4 + 255) / 256, 256, 0, stream>>>(Wo, Woh, nW4);

  // 2) Q/K/V projections: [4096,1024] x [1024,1024]^T
  dim3 pg(B_ * S_ / 16, HID_ / 256);
  proj_gemm<<<pg, 128, 0, stream>>>(xh, Wqh, bq, Qt, nullptr, B_ * S_, HID_, HID_);
  proj_gemm<<<pg, 128, 0, stream>>>(xh, Wkh, bk, Kt, nullptr, B_ * S_, HID_, HID_);
  proj_gemm<<<pg, 128, 0, stream>>>(xh, Wvh, bv, Vt, nullptr, B_ * S_, HID_, HID_);

  // 3) V -> per-head V^T
  v_transpose<<<((int)nX + 255) / 256, 256, 0, stream>>>(Vt, VTh);

  // 4) scores + causal softmax (128 KB dynamic LDS per WG)
  const int lds = 16 * S_ * (int)sizeof(float);
  hipFuncSetAttribute(reinterpret_cast<const void*>(scores_softmax),
                      hipFuncAttributeMaxDynamicSharedMemorySize, lds);
  scores_softmax<<<dim3(S_ / 16, NH_, B_), 128, lds, stream>>>(Qt, Kt, attn);

  // 5) ctx = attn @ V
  ctx_gemm<<<dim3(S_ / 16, NH_, B_), 128, 0, stream>>>(attn, VTh, ctxh);

  // 6) out = ctx @ Wo^T + bo (fp32 output)
  proj_gemm<<<pg, 128, 0, stream>>>(ctxh, Woh, bo, nullptr, out, B_ * S_, HID_, HID_);
}
